// WriteAngleParameters_12790412607611
// MI455X (gfx1250) — matrix operand order repaired and checked by reading, not compiled, not run on hardware
//
#include <hip/hip_runtime.h>
#include <hip/hip_bf16.h>

typedef _Float16 v16h  __attribute__((ext_vector_type(16)));
typedef _Float16 half8 __attribute__((ext_vector_type(8)));
typedef float    v8f   __attribute__((ext_vector_type(8)));

#define NTHREADS 256
#define BLOCK_T  256   // 8 waves * 32 triplets (M=32 per wave, two 16-row tiles)

// ---- weight image / LDS layout (bytes). Fragments are 1 KB: 32 lanes x 32 B. ----
#define OFF_W0   0                      // 24 kc * 4 nt * 1024 = 98304
#define OFF_WS   98304                  // 2*4*1024 = 8192
#define OFF_W1   106496
#define OFF_W2   114688
#define OFF_W3   122880
#define OFF_B0   131072                 // 64 f32
#define OFF_B1   131328
#define OFF_BS2  131584                 // bs[n] + b2[n]
#define OFF_B3   131840
#define OFF_W4   132096                 // stored column-major: w4c[c*64+n]
#define OFF_B4   132608                 // 2 f32
#define WIMG_B   132864                 // total weight-image bytes (16B multiple)

#define SCR_STR  72                     // scratch row stride in halfs (144 B)
#define SCR_HALF 2304                   // 32 rows * 72 halfs (one buffer)
#define SCR_WAVE 9216                   // per-wave scratch bytes (scrA | scrB)
#define OFF_SCR  WIMG_B
#define SMEM_TOTAL (WIMG_B + 8 * SCR_WAVE)   // 206592 B (< 320 KB WGP LDS)

#define PI_F       3.14159265358979f
#define EQ_OVER_PI 0.0318309886183791f  // 0.1 / pi
#define K_STD_F    50.0f
#define K_MOVER_S  2.4f                 // 120/50

static __device__ __forceinline__ float eluf(float x) {
    return x > 0.0f ? x : (__expf(x) - 1.0f);
}

static __device__ __forceinline__ v8f wmma16(v16h a, v16h b, v8f c) {
    // v_wmma_f32_16x16x32_f16
    return __builtin_amdgcn_wmma_f32_16x16x32_f16(false, a, false, b, (short)0, c, false, false);
}

// B-frag layout (16x16x32 f16): lane L holds col n = L%16;
// element s = 2r + p holds K = (s & 8 ? 16 : 0) + (L>=16 ? 8 : 0) + (s & 7). Byte = lane*32 + r*4 + p*2.
static __device__ __forceinline__ int bfrag_byte(int k, int n) {
    int kl = k & 31;
    int lane = (n & 15) + ((kl & 8) ? 16 : 0);
    int s = (kl & 7) + ((kl & 16) ? 8 : 0);
    int r = s >> 1, p = s & 1;
    return lane * 32 + r * 4 + p * 2;
}

static __device__ __forceinline__ void stage_w(const float* __restrict__ W, char* dst,
                                               int offBase, int K, int tid) {
    for (int e = tid; e < K * 64; e += NTHREADS) {
        int k = e >> 6, n = e & 63;
        int kc = k >> 5, nt = (n >> 4);
        int off = offBase + (((kc << 2) + nt) << 10) + bfrag_byte(k, n);
        *(_Float16*)(dst + off) = (_Float16)W[e];
    }
}

static __device__ __forceinline__ void stage_small(char* dst, int tid,
                                                   const float* b0, const float* b1,
                                                   const float* bs, const float* b2,
                                                   const float* b3, const float* W4,
                                                   const float* b4) {
    if (tid < 64) {
        ((float*)(dst + OFF_B0))[tid]  = b0[tid];
        ((float*)(dst + OFF_B1))[tid]  = b1[tid];
        ((float*)(dst + OFF_BS2))[tid] = bs[tid] + b2[tid];
        ((float*)(dst + OFF_B3))[tid]  = b3[tid];
    }
    if (tid < 128) {                    // column-major W4: w4c[c*64+n]
        int n = tid >> 1, c = tid & 1;
        ((float*)(dst + OFF_W4))[c * 64 + n] = W4[n * 2 + c];
    }
    if (tid < 2) ((float*)(dst + OFF_B4))[tid] = b4[tid];
}

// Load an A-fragment (16 rows x 32 K, f16) from [rows][SCR_STR] f16 scratch.
static __device__ __forceinline__ v16h load_afrag(const _Float16* scr, int row, int kc, int klo) {
    union { v16h v; half8 g[2]; } u;
    const _Float16* p = scr + row * SCR_STR + kc * 32 + klo;
    u.g[0] = *(const half8*)(p);        // K = kc*32 + klo + 0..7
    u.g[1] = *(const half8*)(p + 16);   // K = kc*32 + 16 + klo + 0..7
    return u.v;
}

static __device__ __forceinline__ v16h load_bfrag(const char* smem, int offBase,
                                                  int kc, int nt, int lane) {
    return *(const v16h*)(smem + offBase + (((kc << 2) + nt) << 10) + lane * 32);
}

// Gather one A-fragment (this lane's slice) from global h.
template <bool PRE>
static __device__ __forceinline__ v16h load_a_global(const void* hsrc, int idx,
                                                     int cb, int klo) {
    union { v16h v; half8 g[2]; } a;
    if constexpr (PRE) {
        const _Float16* hr = (const _Float16*)hsrc + (size_t)idx * 256;
        a.g[0] = *(const half8*)(hr + cb + klo);
        a.g[1] = *(const half8*)(hr + cb + 16 + klo);
    } else {
        const float* hr = (const float*)hsrc + (size_t)idx * 256;
        const float* src = hr + cb + klo;
        float4 f0 = *(const float4*)(src);
        float4 f1 = *(const float4*)(src + 4);
        float4 f2 = *(const float4*)(src + 16);
        float4 f3 = *(const float4*)(src + 20);
        a.v[0] = (_Float16)f0.x;  a.v[1] = (_Float16)f0.y;
        a.v[2] = (_Float16)f0.z;  a.v[3] = (_Float16)f0.w;
        a.v[4] = (_Float16)f1.x;  a.v[5] = (_Float16)f1.y;
        a.v[6] = (_Float16)f1.z;  a.v[7] = (_Float16)f1.w;
        a.v[8]  = (_Float16)f2.x; a.v[9]  = (_Float16)f2.y;
        a.v[10] = (_Float16)f2.z; a.v[11] = (_Float16)f2.w;
        a.v[12] = (_Float16)f3.x; a.v[13] = (_Float16)f3.y;
        a.v[14] = (_Float16)f3.z; a.v[15] = (_Float16)f3.w;
    }
    return a.v;
}

// ELU(+bias) epilogue: C-layout accumulators -> f16 scratch rows [tileOff..tileOff+15].
static __device__ __forceinline__ void epilogue_store(_Float16* scr, int tileOff,
                                                      int hi, int nl,
                                                      const v8f* acc, const float* bn) {
#pragma unroll
    for (int nt = 0; nt < 4; ++nt)
#pragma unroll
        for (int v = 0; v < 8; ++v)
            scr[(tileOff + v + hi * 8) * SCR_STR + nt * 16 + nl] =
                (_Float16)eluf(acc[nt][v] + bn[nt]);
}

// ---- pre-pass 1: h (f32) -> h16 (f16), 8 elements per thread ----
extern "C" __global__ void __launch_bounds__(NTHREADS)
cvt_h_kernel(const float* __restrict__ h, _Float16* __restrict__ h16, long n8) {
    long g = (long)blockIdx.x * NTHREADS + threadIdx.x;
    if (g >= n8) return;
    const float* s = h + g * 8;
    float4 f0 = *(const float4*)(s);
    float4 f1 = *(const float4*)(s + 4);
    half8 o;
    o[0] = (_Float16)f0.x; o[1] = (_Float16)f0.y;
    o[2] = (_Float16)f0.z; o[3] = (_Float16)f0.w;
    o[4] = (_Float16)f1.x; o[5] = (_Float16)f1.y;
    o[6] = (_Float16)f1.z; o[7] = (_Float16)f1.w;
    *(half8*)(h16 + g * 8) = o;
}

// ---- pre-pass 2: build the f16 fragment image + biases in global scratch ----
extern "C" __global__ void __launch_bounds__(NTHREADS)
build_wimg_kernel(const float* __restrict__ W0, const float* __restrict__ b0,
                  const float* __restrict__ Ws, const float* __restrict__ bs,
                  const float* __restrict__ W1, const float* __restrict__ b1,
                  const float* __restrict__ W2, const float* __restrict__ b2,
                  const float* __restrict__ W3, const float* __restrict__ b3,
                  const float* __restrict__ W4, const float* __restrict__ b4,
                  char* __restrict__ wimg) {
    const int tid = threadIdx.x;
    stage_w(W0, wimg, OFF_W0, 768, tid);
    stage_w(Ws, wimg, OFF_WS, 64, tid);
    stage_w(W1, wimg, OFF_W1, 64, tid);
    stage_w(W2, wimg, OFF_W2, 64, tid);
    stage_w(W3, wimg, OFF_W3, 64, tid);
    stage_small(wimg, tid, b0, b1, bs, b2, b3, W4, b4);
}

// ---- main kernel. PRE=true: f16 gathers + prebuilt weight image in d_ws. ----
template <bool PRE>
__global__ void __launch_bounds__(NTHREADS)
angle_mlp_kernel(const void* __restrict__ hsrc, const int* __restrict__ idxs,
                 const char* __restrict__ wimg,
                 const float* __restrict__ W0, const float* __restrict__ b0,
                 const float* __restrict__ Ws, const float* __restrict__ bs,
                 const float* __restrict__ W1, const float* __restrict__ b1,
                 const float* __restrict__ W2, const float* __restrict__ b2,
                 const float* __restrict__ W3, const float* __restrict__ b3,
                 const float* __restrict__ W4, const float* __restrict__ b4,
                 float* __restrict__ out, int T) {
    extern __shared__ char smem[];
    const int tid = threadIdx.x;

    if constexpr (PRE) {
        const float4* src = (const float4*)wimg;   // pure b128 copy of the image
        float4* dst = (float4*)smem;
        for (int i = tid; i < WIMG_B / 16; i += NTHREADS) dst[i] = src[i];
    } else {
        stage_w(W0, smem, OFF_W0, 768, tid);
        stage_w(Ws, smem, OFF_WS, 64, tid);
        stage_w(W1, smem, OFF_W1, 64, tid);
        stage_w(W2, smem, OFF_W2, 64, tid);
        stage_w(W3, smem, OFF_W3, 64, tid);
        stage_small(smem, tid, b0, b1, bs, b2, b3, W4, b4);
    }
    __syncthreads();

    const int wave = tid >> 5;
    const int lane = tid & 31;
    const int nl   = lane & 15;
    const int hi   = lane >> 4;
    const int klo  = hi * 8;

    _Float16* scrA = (_Float16*)(smem + OFF_SCR + wave * SCR_WAVE);
    _Float16* scrB = scrA + SCR_HALF;

    const float* b0l  = (const float*)(smem + OFF_B0);
    const float* b1l  = (const float*)(smem + OFF_B1);
    const float* bs2l = (const float*)(smem + OFF_BS2);
    const float* b3l  = (const float*)(smem + OFF_B3);
    const float* w4l  = (const float*)(smem + OFF_W4);
    const float* b4l  = (const float*)(smem + OFF_B4);

    const int tbase = blockIdx.x * BLOCK_T + wave * 32;
    const int t0 = tbase + nl;           // tile 0 row
    const int t1 = t0 + 16;              // tile 1 row
    const int tc0 = t0 < T ? t0 : (T - 1);
    const int tc1 = t1 < T ? t1 : (T - 1);
    const int iA0 = idxs[tc0 * 3 + 0], iA1 = idxs[tc0 * 3 + 1], iA2 = idxs[tc0 * 3 + 2];
    const int iB0 = idxs[tc1 * 3 + 0], iB1 = idxs[tc1 * 3 + 1], iB2 = idxs[tc1 * 3 + 2];

    float bn[4];

    // ---- layer 0: z0 = elu(x @ W0 + b0), K = 768 = 24 chunks of 32 ----
    v8f acc0[4], acc1[4];
#pragma unroll
    for (int nt = 0; nt < 4; ++nt) { acc0[nt] = (v8f){}; acc1[nt] = (v8f){}; }
#pragma unroll
    for (int kc = 0; kc < 24; ++kc) {
        const int j = kc >> 3;
        const int cb = (kc & 7) << 5;
        const int ia = (j == 0) ? iA0 : ((j == 1) ? iA1 : iA2);
        const int ib = (j == 0) ? iB0 : ((j == 1) ? iB1 : iB2);
        v16h a0 = load_a_global<PRE>(hsrc, ia, cb, klo);
        v16h a1 = load_a_global<PRE>(hsrc, ib, cb, klo);
#pragma unroll
        for (int nt = 0; nt < 4; ++nt) {
            v16h b = load_bfrag(smem, OFF_W0, kc, nt, lane);   // shared by both tiles
            acc0[nt] = wmma16(a0, b, acc0[nt]);
            acc1[nt] = wmma16(a1, b, acc1[nt]);
        }
    }
#pragma unroll
    for (int nt = 0; nt < 4; ++nt) bn[nt] = b0l[nt * 16 + nl];
    epilogue_store(scrA, 0, hi, nl, acc0, bn);
    epilogue_store(scrA, 16, hi, nl, acc1, bn);
    __syncthreads();

    // ---- residual block: S = z0@Ws + (bs+b2) + elu(z0@W1+b1)@W2 ----
    v8f accS0[4], accS1[4], accT0[4], accT1[4];
#pragma unroll
    for (int nt = 0; nt < 4; ++nt) {
        accS0[nt] = (v8f){}; accS1[nt] = (v8f){};
        accT0[nt] = (v8f){}; accT1[nt] = (v8f){};
    }
#pragma unroll
    for (int kc = 0; kc < 2; ++kc) {
        v16h a0 = load_afrag(scrA, nl, kc, klo);
        v16h a1 = load_afrag(scrA + 16 * SCR_STR, nl, kc, klo);
#pragma unroll
        for (int nt = 0; nt < 4; ++nt) {
            v16h bS = load_bfrag(smem, OFF_WS, kc, nt, lane);
            accS0[nt] = wmma16(a0, bS, accS0[nt]);
            accS1[nt] = wmma16(a1, bS, accS1[nt]);
            v16h bT = load_bfrag(smem, OFF_W1, kc, nt, lane);
            accT0[nt] = wmma16(a0, bT, accT0[nt]);
            accT1[nt] = wmma16(a1, bT, accT1[nt]);
        }
    }
#pragma unroll
    for (int nt = 0; nt < 4; ++nt) bn[nt] = b1l[nt * 16 + nl];
    epilogue_store(scrB, 0, hi, nl, accT0, bn);
    epilogue_store(scrB, 16, hi, nl, accT1, bn);
    __syncthreads();
#pragma unroll
    for (int kc = 0; kc < 2; ++kc) {
        v16h a0 = load_afrag(scrB, nl, kc, klo);
        v16h a1 = load_afrag(scrB + 16 * SCR_STR, nl, kc, klo);
#pragma unroll
        for (int nt = 0; nt < 4; ++nt) {
            v16h b = load_bfrag(smem, OFF_W2, kc, nt, lane);
            accS0[nt] = wmma16(a0, b, accS0[nt]);
            accS1[nt] = wmma16(a1, b, accS1[nt]);
        }
    }
#pragma unroll
    for (int nt = 0; nt < 4; ++nt) bn[nt] = bs2l[nt * 16 + nl];
    epilogue_store(scrA, 0, hi, nl, accS0, bn);
    epilogue_store(scrA, 16, hi, nl, accS1, bn);
    __syncthreads();

    // ---- layer 3: z3 = elu(z @ W3 + b3) ----
    v8f accU0[4], accU1[4];
#pragma unroll
    for (int nt = 0; nt < 4; ++nt) { accU0[nt] = (v8f){}; accU1[nt] = (v8f){}; }
#pragma unroll
    for (int kc = 0; kc < 2; ++kc) {
        v16h a0 = load_afrag(scrA, nl, kc, klo);
        v16h a1 = load_afrag(scrA + 16 * SCR_STR, nl, kc, klo);
#pragma unroll
        for (int nt = 0; nt < 4; ++nt) {
            v16h b = load_bfrag(smem, OFF_W3, kc, nt, lane);
            accU0[nt] = wmma16(a0, b, accU0[nt]);
            accU1[nt] = wmma16(a1, b, accU1[nt]);
        }
    }
#pragma unroll
    for (int nt = 0; nt < 4; ++nt) bn[nt] = b3l[nt * 16 + nl];
    epilogue_store(scrA, 0, hi, nl, accU0, bn);
    epilogue_store(scrA, 16, hi, nl, accU1, bn);
    __syncthreads();

    // ---- head: coeffs = z3 @ W4 + b4; lane = (row=nl, component=hi), 2 tiles ----
    const float* w4c = w4l + hi * 64;
#pragma unroll
    for (int tile = 0; tile < 2; ++tile) {
        const _Float16* zr = scrA + (tile * 16 + nl) * SCR_STR;
        float sum = b4l[hi];
#pragma unroll
        for (int n = 0; n < 64; ++n)
            sum += (float)zr[n] * w4c[n];
        float res;
        if (hi == 0) {
            res = PI_F / (1.0f + __expf(-sum * EQ_OVER_PI));   // pi * sigmoid(c0*0.1/pi)
        } else {
            float x = sum + K_MOVER_S;                          // 50 * elu(c1 + 2.4)
            res = K_STD_F * (x > 0.0f ? x : (__expf(x) - 1.0f));
        }
        const int t = tbase + tile * 16 + nl;
        if (t < T) out[t * 2 + hi] = res;
    }
}

extern "C" void kernel_launch(void* const* d_in, const int* in_sizes, int n_in,
                              void* d_out, int out_size, void* d_ws, size_t ws_size,
                              hipStream_t stream) {
    const float* h  = (const float*)d_in[0];
    const int* idxs = (const int*)d_in[1];
    const float* W0 = (const float*)d_in[2];
    const float* b0 = (const float*)d_in[3];
    const float* Ws = (const float*)d_in[4];
    const float* bs = (const float*)d_in[5];
    const float* W1 = (const float*)d_in[6];
    const float* b1 = (const float*)d_in[7];
    const float* W2 = (const float*)d_in[8];
    const float* b2 = (const float*)d_in[9];
    const float* W3 = (const float*)d_in[10];
    const float* b3 = (const float*)d_in[11];
    const float* W4 = (const float*)d_in[12];
    const float* b4 = (const float*)d_in[13];
    const int T = in_sizes[1] / 3;
    const int blocks = (T + BLOCK_T - 1) / BLOCK_T;

    const long   hElems = (long)in_sizes[0];
    const size_t h16B   = (size_t)hElems * 2;         // f16 copy of h
    const size_t need   = h16B + WIMG_B;

    if (ws_size >= need) {
        _Float16* h16 = (_Float16*)d_ws;
        char* wimg = (char*)d_ws + h16B;
        const long n8 = hElems / 8;
        const int cblocks = (int)((n8 + NTHREADS - 1) / NTHREADS);
        cvt_h_kernel<<<cblocks, NTHREADS, 0, stream>>>(h, h16, n8);
        build_wimg_kernel<<<1, NTHREADS, 0, stream>>>(W0, b0, Ws, bs, W1, b1,
                                                      W2, b2, W3, b3, W4, b4, wimg);
        angle_mlp_kernel<true><<<blocks, NTHREADS, SMEM_TOTAL, stream>>>(
            (const void*)h16, idxs, (const char*)wimg,
            W0, b0, Ws, bs, W1, b1, W2, b2, W3, b3, W4, b4,
            (float*)d_out, T);
    } else {
        angle_mlp_kernel<false><<<blocks, NTHREADS, SMEM_TOTAL, stream>>>(
            (const void*)h, idxs, (const char*)nullptr,
            W0, b0, Ws, bs, W1, b1, W2, b2, W3, b3, W4, b4,
            (float*)d_out, T);
    }
}
